// svdconv_66632122630499
// MI455X (gfx1250) — matrix operand order repaired and checked by reading, not compiled, not run on hardware
//
#include <hip/hip_runtime.h>
#include <hip/hip_bf16.h>
#include <stdint.h>

// ---------------------------------------------------------------------------
// SVD-factorized 3x3 conv:  out[b,o,h,w] = sum_t U[t] @ (V[t] @ x_shift(t))
// x:(32,256,56,56) f32, U:(9,256,64), V:(9,64,256), out:(32,256,56,56)
// Strategy: pack x -> padded NHWC bf16 (55MB, L2-resident), then fused
// dual-GEMM per output row using v_wmma_f32_16x16x32_bf16.
// Round 2: stage each padded x row ONCE per tap-row i (j-shift folded into
// LDS read addresses), and fill LDS with global_load_async_to_lds_b128
// (ASYNCcnt-tracked DMA) instead of VGPR round-trips.
// ---------------------------------------------------------------------------

typedef __attribute__((ext_vector_type(16))) __bf16    v16bf;
typedef __attribute__((ext_vector_type(8)))  float     v8f;
typedef __attribute__((ext_vector_type(4)))  uint32_t  u32x4;

#define B_    32
#define CIN   256
#define HH    56
#define WW    56
#define RR    64
#define COUT  256
#define HP    58
#define WP    58
#define NTILE 64    // padded pixel tile (one output row)
#define XROWS 66    // staged rows incl. j-shift overread slack

static constexpr size_t XP_ELEMS = (size_t)B_ * HP * WP * CIN;   // 27,557,888
static constexpr size_t UV_ELEMS = (size_t)9 * COUT * RR;        // 147,456
static constexpr size_t U_OFF    = XP_ELEMS;                     // elem offsets in ws
static constexpr size_t V_OFF    = XP_ELEMS + UV_ELEMS;

union Frag16 { v16bf bf; u32x4 u[2]; };
union PackBF2 { __bf16 h[2]; uint32_t u; };

// LDS byte offset of a generic pointer into a __shared__ object
__device__ __forceinline__ uint32_t lds_off(const void* p) {
    return (uint32_t)(uintptr_t)(const __attribute__((address_space(3))) char*)p;
}

// ---- pack x (f32 BCHW) -> xp (bf16, [b][hp][wp][c], zero pad border) ------
__global__ void svdconv_pack_x(const float* __restrict__ x, __bf16* __restrict__ xp) {
    size_t idx = (size_t)blockIdx.x * blockDim.x + threadIdx.x;
    if (idx >= XP_ELEMS) return;
    int c    = (int)(idx & (CIN - 1));
    size_t p = idx >> 8;                    // (b, hp, wp)
    int wp   = (int)(p % WP);
    int hp   = (int)((p / WP) % HP);
    int b    = (int)(p / (WP * HP));
    int h = hp - 1, w = wp - 1;
    float v = 0.0f;
    if (h >= 0 && h < HH && w >= 0 && w < WW)
        v = x[(((size_t)b * CIN + c) * HH + h) * WW + w];
    xp[idx] = (__bf16)v;
}

// ---- pack U (9,256,64) and V (9,64,256) to bf16 (row-major preserved) -----
__global__ void svdconv_pack_uv(const float* __restrict__ U, const float* __restrict__ V,
                                __bf16* __restrict__ Ub, __bf16* __restrict__ Vb) {
    int idx = blockIdx.x * blockDim.x + threadIdx.x;
    if (idx < (int)UV_ELEMS) {
        Ub[idx] = (__bf16)U[idx];
        Vb[idx] = (__bf16)V[idx];
    }
}

// ---- fused dual-GEMM conv kernel ------------------------------------------
// grid: B*HH workgroups, 256 threads (8 wave32). Each block: one output row
// (64-pixel padded N tile, 56 valid) x all 256 output channels.
__global__ __launch_bounds__(256)
void svdconv_main(const __bf16* __restrict__ xp,
                  const __bf16* __restrict__ Ub,
                  const __bf16* __restrict__ Vb,
                  float* __restrict__ out) {
    __shared__ __align__(16) __bf16 lds_x[XROWS * CIN];   // [wp][c]  33 KB
    __shared__ __align__(16) __bf16 lds_mid[NTILE * RR];  // [n][r]    8 KB

    const int tid  = threadIdx.x;
    const int lane = tid & 31;
    const int wave = tid >> 5;
    const int lm   = lane & 15;    // column / row-in-16
    const int lh   = lane >> 4;    // K-half / M-half selector

    const int b = blockIdx.x / HH;
    const int h = blockIdx.x % HH;

    // GEMM1 tile assignment: wave -> r-tile rt (0..3), n-tile pair ntp
    const int rt  = wave & 3;
    const int ntp = (wave >> 2) * 2;
    // GEMM2 tile assignment: wave -> o-tiles {2w, 2w+1} x n-tiles 0..3
    const int ot0 = wave * 2;

    v8f acc[2][4] = {};   // 256x64 output tile, distributed

    const int nfill = tid & 63;   // LDS fill: padded-row pixel slot
    const int pfill = tid >> 6;   // LDS fill: 64-channel chunk (0..3)

    for (int i = 0; i < 3; ++i) {
        // ---- stage padded x row hp = h+i once: lds_x[wp][c], async DMA ----
        {
            const int hp = h + i;
            const int wp = (nfill > WP - 1) ? (WP - 1) : nfill;  // rows>=58 dead
            const __bf16* g = xp + (((size_t)b * HP + hp) * WP + wp) * CIN + pfill * 64;
            const uint32_t l = lds_off(lds_x + nfill * CIN + pfill * 64);
            #pragma unroll
            for (int q = 0; q < 8; ++q) {
                asm volatile("global_load_async_to_lds_b128 %0, %1, off"
                             :: "v"(l + q * 16),
                                "v"((unsigned long long)(uintptr_t)(g + q * 8))
                             : "memory");
            }
        }
        asm volatile("s_wait_asynccnt 0x0" ::: "memory");
        __syncthreads();

        for (int j = 0; j < 3; ++j) {
            const int t = i * 3 + j;

            // ---- GEMM1: mid(64x64) = V[t](64x256) x xT(256x64) ------------
            // j-shift: output pixel n reads staged row slot wp = n + j
            v8f macc[2] = {};
            const __bf16* Vrow = Vb + ((size_t)t * RR + (rt * 16 + lm)) * CIN;
            #pragma unroll
            for (int ks = 0; ks < 8; ++ks) {
                const int kb = ks * 32;
                Frag16 a;   // A layout: m=lm, VGPR v: K = lh*16 + 2v (+1)
                a.u[0] = *(const u32x4*)(Vrow + kb + lh * 16);
                a.u[1] = *(const u32x4*)(Vrow + kb + lh * 16 + 8);
                #pragma unroll
                for (int q = 0; q < 2; ++q) {
                    const int n = (ntp + q) * 16 + lm;
                    Frag16 bf;  // B layout: n=lm, VGPR v: K = lh*16 + 2v (+1)
                    const u32x4* bp = (const u32x4*)(lds_x + (n + j) * CIN + kb + lh * 16);
                    bf.u[0] = bp[0]; bf.u[1] = bp[1];
                    macc[q] = __builtin_amdgcn_wmma_f32_16x16x32_bf16(
                        false, a.bf, false, bf.bf, (short)0, macc[q], false, false);
                }
            }
            // mid -> LDS [n][r] as bf16 (C/D layout: m = lh*8 + v, n = lm)
            #pragma unroll
            for (int q = 0; q < 2; ++q) {
                const int n = (ntp + q) * 16 + lm;
                const int rbase = rt * 16 + lh * 8;
                #pragma unroll
                for (int v = 0; v < 8; v += 2) {
                    PackBF2 pk;
                    pk.h[0] = (__bf16)macc[q][v];
                    pk.h[1] = (__bf16)macc[q][v + 1];
                    *(uint32_t*)(lds_mid + n * RR + rbase + v) = pk.u;
                }
            }
            __syncthreads();

            // ---- GEMM2: out(256x64) += U[t](256x64) x mid(64x64) ----------
            #pragma unroll
            for (int ks = 0; ks < 2; ++ks) {
                const int kb = ks * 32;
                Frag16 bf[4];
                #pragma unroll
                for (int nt = 0; nt < 4; ++nt) {
                    const int n = nt * 16 + lm;
                    const u32x4* bp = (const u32x4*)(lds_mid + n * RR + kb + lh * 16);
                    bf[nt].u[0] = bp[0]; bf[nt].u[1] = bp[1];
                }
                #pragma unroll
                for (int oi = 0; oi < 2; ++oi) {
                    const int o = (ot0 + oi) * 16 + lm;
                    const __bf16* Urow = Ub + ((size_t)t * COUT + o) * RR;
                    Frag16 a;
                    a.u[0] = *(const u32x4*)(Urow + kb + lh * 16);
                    a.u[1] = *(const u32x4*)(Urow + kb + lh * 16 + 8);
                    #pragma unroll
                    for (int nt = 0; nt < 4; ++nt)
                        acc[oi][nt] = __builtin_amdgcn_wmma_f32_16x16x32_bf16(
                            false, a.bf, false, bf[nt].bf, (short)0, acc[oi][nt], false, false);
                }
            }
            __syncthreads();   // mid consumed; safe to overwrite next tap
        }

        // prefetch next tap-row of x toward L2
        if (i < 2) {
            __builtin_prefetch(xp + (((size_t)b * HP + h + i + 1) * WP) * CIN, 0, 1);
        }
    }

    // ---- write out[b][o][h][w] (f32), mask dead columns -------------------
    #pragma unroll
    for (int oi = 0; oi < 2; ++oi) {
        #pragma unroll
        for (int nt = 0; nt < 4; ++nt) {
            const int n = nt * 16 + lm;
            if (n < WW) {
                #pragma unroll
                for (int v = 0; v < 8; ++v) {
                    const int o = (ot0 + oi) * 16 + lh * 8 + v;
                    out[(((size_t)b * COUT + o) * HH + h) * WW + n] = acc[oi][nt][v];
                }
            }
        }
    }
}

// ---------------------------------------------------------------------------
extern "C" void kernel_launch(void* const* d_in, const int* in_sizes, int n_in,
                              void* d_out, int out_size, void* d_ws, size_t ws_size,
                              hipStream_t stream) {
    const float* x = (const float*)d_in[0];
    const float* U = (const float*)d_in[1];
    const float* V = (const float*)d_in[2];
    float* out = (float*)d_out;

    __bf16* ws = (__bf16*)d_ws;           // needs ~55.7 MB of ws
    __bf16* xp = ws;
    __bf16* Ubf = ws + U_OFF;
    __bf16* Vbf = ws + V_OFF;

    {
        const int threads = 256;
        const unsigned blocks = (unsigned)((XP_ELEMS + threads - 1) / threads);
        svdconv_pack_x<<<dim3(blocks), dim3(threads), 0, stream>>>(x, xp);
    }
    svdconv_pack_uv<<<dim3((unsigned)((UV_ELEMS + 255) / 256)), dim3(256), 0, stream>>>(U, V, Ubf, Vbf);
    svdconv_main<<<dim3(B_ * HH), dim3(256), 0, stream>>>(xp, Ubf, Vbf, out);
}
